// MoE_33621003993134
// MI455X (gfx1250) — compile-verified, matrix operand order checked by reading, not run on hardware
//
#include <hip/hip_runtime.h>

#define BT_   8192   // B*T
#define DD_   1024   // D
#define HH_   4096   // H
#define EE_   8      // experts

typedef __attribute__((ext_vector_type(16))) __bf16 v16bf;
typedef __attribute__((ext_vector_type(8)))  float  v8f;

// ---------------- workspace layout (bytes) ----------------
// counts : 8 ints                     @ 0
// list   : E*BT ints                  @ 1024
// W1t    : bf16 [E][H][D]             @ W1T_OFF
// W2t    : bf16 [E][D][H]             @ W2T_OFF
// hbuf   : bf16 [BT][H]               @ HBUF_OFF
static const size_t LIST_OFF = 1024;
static const size_t W1T_OFF  = LIST_OFF + (size_t)EE_ * BT_ * 4;
static const size_t W2T_OFF  = W1T_OFF + (size_t)EE_ * HH_ * DD_ * 2;
static const size_t HBUF_OFF = W2T_OFF + (size_t)EE_ * DD_ * HH_ * 2;

// -----------------------------------------------------------------
// CDNA5 async global->LDS copy (ASYNCcnt-tracked), 16 bytes per lane.
// VDST holds the per-lane LDS byte offset, VADDR the 64-bit global address.
__device__ __forceinline__ void moe_async_ld_b128(uint32_t lds_off, const void* g) {
    asm volatile("global_load_async_to_lds_b128 %0, %1, off"
                 :: "v"(lds_off), "v"((unsigned long long)(uintptr_t)g)
                 : "memory");
}
__device__ __forceinline__ void moe_wait_async0() {
    asm volatile("s_wait_asynccnt 0x0" ::: "memory");
}
// Generic shared pointer -> LDS byte offset (LDS aperture keeps offset in low dword).
__device__ __forceinline__ uint32_t moe_lds_off(const void* p) {
    return (uint32_t)(uintptr_t)p;
}

// -----------------------------------------------------------------
__global__ void moe_zero_counts(int* __restrict__ counts) {
    if (threadIdx.x < EE_) counts[threadIdx.x] = 0;
}

// in: [E][R][C] f32  ->  out: [E][C][R] bf16   (LDS-tiled transpose+convert)
__global__ __launch_bounds__(256) void moe_cvt_transpose(
        const float* __restrict__ in, __bf16* __restrict__ out, int R, int C) {
    __shared__ __bf16 tile[32][33];
    const int e  = blockIdx.z;
    const int r0 = blockIdx.y * 32;
    const int c0 = blockIdx.x * 32;
    const int tx = threadIdx.x & 31;
    const int ty = threadIdx.x >> 5;
    const float* src = in  + (size_t)e * R * C;
    __bf16*      dst = out + (size_t)e * C * R;
    #pragma unroll
    for (int i = ty; i < 32; i += 8)
        tile[i][tx] = (__bf16)src[(size_t)(r0 + i) * C + (c0 + tx)];
    __syncthreads();
    #pragma unroll
    for (int i = ty; i < 32; i += 8)
        dst[(size_t)(c0 + i) * R + (r0 + tx)] = tile[tx][i];
}

// -----------------------------------------------------------------
// Router: one wave (32 lanes) per token. Wr cached in LDS.
__global__ __launch_bounds__(256) void moe_router(
        const float* __restrict__ x, const float* __restrict__ Wr,
        const float* __restrict__ br, float* __restrict__ probs,
        int* __restrict__ eidx, int* __restrict__ counts, int* __restrict__ list) {
    __shared__ float sWr[DD_ * EE_];
    const int tid = threadIdx.x;
    for (int i = tid; i < DD_ * EE_; i += 256) sWr[i] = Wr[i];
    __syncthreads();

    const int lane  = tid & 31;
    const int token = blockIdx.x * 8 + (tid >> 5);
    const float* xr = x + (size_t)token * DD_;

    float acc[EE_] = {0.f, 0.f, 0.f, 0.f, 0.f, 0.f, 0.f, 0.f};
    for (int d = lane; d < DD_; d += 32) {
        const float xv = xr[d];
        #pragma unroll
        for (int e = 0; e < EE_; ++e) acc[e] = fmaf(xv, sWr[d * EE_ + e], acc[e]);
    }
    #pragma unroll
    for (int off = 16; off > 0; off >>= 1) {
        #pragma unroll
        for (int e = 0; e < EE_; ++e) acc[e] += __shfl_down(acc[e], off, 32);
    }
    if (lane == 0) {
        float l[EE_], p[EE_];
        float m = -3.4e38f; int best = 0;
        #pragma unroll
        for (int e = 0; e < EE_; ++e) {
            l[e] = acc[e] + br[e];
            if (l[e] > m) { m = l[e]; best = e; }   // first-max tie rule (jnp.argmax)
        }
        float s = 0.f;
        #pragma unroll
        for (int e = 0; e < EE_; ++e) { p[e] = expf(l[e] - m); s += p[e]; }
        const float inv = 1.f / s;
        #pragma unroll
        for (int e = 0; e < EE_; ++e) probs[(size_t)token * EE_ + e] = p[e] * inv;
        eidx[token] = best;
        const int pos = atomicAdd(&counts[best], 1);
        list[best * BT_ + pos] = token;
    }
}

// -----------------------------------------------------------------
// Load one 16x16x32 bf16 operand fragment from an LDS tile stored
// row-major [row][k] with pitch 40 bf16 (80B = 5 x 16B, conflict-spread).
// 16-bit A layout (ISA 7.12.2): lane half h holds K = 8h+{0..7}, 16+8h+{0..7}.
__device__ __forceinline__ v16bf moe_load_frag(const __bf16* tile, int rowBase, int lane) {
    const int half = lane >> 4;
    const int r    = rowBase + (lane & 15);
    const uint32_t* p = (const uint32_t*)(tile + r * 40);
    const int w0 = half * 4;                 // dword index of k = 8*half
    union { uint32_t u[8]; v16bf v; } f;
    #pragma unroll
    for (int j = 0; j < 4; ++j) {
        f.u[j]     = p[w0 + j];              // K = 8h .. 8h+7
        f.u[4 + j] = p[8 + w0 + j];          // K = 16+8h .. 16+8h+7
    }
    return f.v;
}

// Grouped GEMM over one expert's token list, double-buffered LDS.
//   FIRST:  A = gathered x rows (f32 -> bf16, register-pipelined), B = W1t (async->LDS),
//           out = relu -> bf16 hbuf
//   !FIRST: A = gathered hbuf rows (async->LDS), B = W2t (async->LDS),
//           out = f32 scatter to d_out
// Block tile 128x128x32, 8 waves (2M x 4N), wave tile 64x32 = 4x2 WMMA tiles.
template<int KTOT, int NTOT, bool FIRST>
__global__ __launch_bounds__(256) void moe_gemm(
        const void* __restrict__ Ain, const __bf16* __restrict__ Bt,
        const float* __restrict__ bias, const int* __restrict__ counts,
        const int* __restrict__ list, void* __restrict__ Outp) {
    __shared__ __bf16 As[2][128 * 40];
    __shared__ __bf16 Bs[2][128 * 40];

    const int e    = blockIdx.z;
    const int cnt  = counts[e];
    const int mblk = blockIdx.y;
    if (mblk * 128 >= cnt) return;           // uniform early-exit
    const int nblk = blockIdx.x;

    const int tid  = threadIdx.x;
    const int lane = tid & 31;
    const int wave = tid >> 5;
    const int wm   = (wave & 1) * 64;        // wave M offset within block tile
    const int wn   = (wave >> 1) * 32;       // wave N offset within block tile

    const int* elist = list + e * BT_;

    // Staging assignment: 2 threads per row, 16 K-elements (32B) each.
    const int lr = tid >> 1;                 // tile row 0..127
    const int lh = tid & 1;                  // which 16-wide half of the K slice
    int apos = mblk * 128 + lr;
    if (apos >= cnt) apos = cnt - 1;         // clamp gather for ragged tail
    const int atok = elist[apos];

    const float*  Arow_f = FIRST ? ((const float*)Ain  + (size_t)atok * KTOT) : nullptr;
    const __bf16* Arow_h = FIRST ? nullptr : ((const __bf16*)Ain + (size_t)atok * KTOT);
    const __bf16* Brow   = Bt + (size_t)e * NTOT * KTOT + (size_t)(nblk * 128 + lr) * KTOT;

    uint32_t as_off[2], bs_off[2];
    #pragma unroll
    for (int b = 0; b < 2; ++b) {
        as_off[b] = moe_lds_off(&As[b][lr * 40 + lh * 16]);
        bs_off[b] = moe_lds_off(&Bs[b][lr * 40 + lh * 16]);
    }

    // -- staging helpers --
    auto stageB = [&](int buf, int k0) {
        const __bf16* g = Brow + k0 + lh * 16;
        moe_async_ld_b128(bs_off[buf],      g);
        moe_async_ld_b128(bs_off[buf] + 16, g + 8);
    };
    auto stageA_async = [&](int buf, int k0) {   // !FIRST path
        const __bf16* g = Arow_h + k0 + lh * 16;
        moe_async_ld_b128(as_off[buf],      g);
        moe_async_ld_b128(as_off[buf] + 16, g + 8);
    };
    struct AF { float4 q0, q1, q2, q3; };
    auto loadA = [&](int k0) {                   // FIRST path: global f32 -> regs
        const float4* s = (const float4*)(Arow_f + k0 + lh * 16);
        AF r; r.q0 = s[0]; r.q1 = s[1]; r.q2 = s[2]; r.q3 = s[3];
        return r;
    };
    auto storeA = [&](int buf, const AF& r) {    // FIRST path: cvt + LDS store
        const float fx[16] = {r.q0.x, r.q0.y, r.q0.z, r.q0.w,
                              r.q1.x, r.q1.y, r.q1.z, r.q1.w,
                              r.q2.x, r.q2.y, r.q2.z, r.q2.w,
                              r.q3.x, r.q3.y, r.q3.z, r.q3.w};
        union { __bf16 h[16]; uint4 q[2]; } t;
        #pragma unroll
        for (int i = 0; i < 16; ++i) t.h[i] = (__bf16)fx[i];
        uint4* d = (uint4*)&As[buf][lr * 40 + lh * 16];
        d[0] = t.q[0]; d[1] = t.q[1];
    };

    v8f acc[4][2];
    #pragma unroll
    for (int mi = 0; mi < 4; ++mi)
        #pragma unroll
        for (int ni = 0; ni < 2; ++ni) { v8f z = {}; acc[mi][ni] = z; }

    // ---- prologue: stage tile 0 into buffer 0 ----
    if constexpr (FIRST) { AF a0 = loadA(0); storeA(0, a0); }
    else                 { stageA_async(0, 0); }
    stageB(0, 0);
    moe_wait_async0();
    __syncthreads();

    // ---- main loop: compute buf, stage buf^1 ----
    for (int k0 = 0; k0 < KTOT; k0 += 32) {
        const int  buf  = (k0 >> 5) & 1;
        const bool more = (k0 + 32) < KTOT;

        AF anext = {};
        if (more) {
            if constexpr (FIRST) anext = loadA(k0 + 32);   // issue loads early
            else                 stageA_async(buf ^ 1, k0 + 32);
            stageB(buf ^ 1, k0 + 32);
            __builtin_prefetch(Brow + k0 + 64 + lh * 16, 0, 1);  // global_prefetch_b8
        }

        // compute from `buf` (hides the staging latency above)
        v16bf a[4], b[2];
        #pragma unroll
        for (int mi = 0; mi < 4; ++mi) a[mi] = moe_load_frag(As[buf], wm + mi * 16, lane);
        #pragma unroll
        for (int ni = 0; ni < 2; ++ni) b[ni] = moe_load_frag(Bs[buf], wn + ni * 16, lane);
        #pragma unroll
        for (int mi = 0; mi < 4; ++mi)
            #pragma unroll
            for (int ni = 0; ni < 2; ++ni)
                acc[mi][ni] = __builtin_amdgcn_wmma_f32_16x16x32_bf16(
                    false, a[mi], false, b[ni], (short)0, acc[mi][ni], false, false);

        if (more) {
            if constexpr (FIRST) storeA(buf ^ 1, anext);   // loadcnt wait lands here
            moe_wait_async0();                             // buf^1 fully in LDS
            __syncthreads();
        }
    }

    // ---- epilogue: bias (+relu->bf16 | f32 scatter) ----
    // C/D layout: VGPR j, lane L: m = j + 8*(L/16), n = L%16.
    const int half  = lane >> 4;
    const int nlane = lane & 15;
    #pragma unroll
    for (int ni = 0; ni < 2; ++ni) {
        const int col = nblk * 128 + wn + ni * 16 + nlane;
        const float bv = bias[(size_t)e * NTOT + col];
        #pragma unroll
        for (int mi = 0; mi < 4; ++mi) {
            #pragma unroll
            for (int j = 0; j < 8; ++j) {
                const int pos = mblk * 128 + wm + mi * 16 + half * 8 + j;
                if (pos < cnt) {
                    const int token = elist[pos];
                    const float v = acc[mi][ni][j] + bv;
                    if constexpr (FIRST)
                        ((__bf16*)Outp)[(size_t)token * NTOT + col] = (__bf16)fmaxf(v, 0.f);
                    else
                        ((float*)Outp)[(size_t)token * NTOT + col] = v;
                }
            }
        }
    }
}

// -----------------------------------------------------------------
extern "C" void kernel_launch(void* const* d_in, const int* in_sizes, int n_in,
                              void* d_out, int out_size, void* d_ws, size_t ws_size,
                              hipStream_t stream) {
    (void)in_sizes; (void)n_in; (void)out_size; (void)ws_size;

    const float* x  = (const float*)d_in[0];   // [B,T,D]
    const float* Wr = (const float*)d_in[1];   // [D,E]
    const float* br = (const float*)d_in[2];   // [E]
    const float* W1 = (const float*)d_in[3];   // [E,D,H]
    const float* b1 = (const float*)d_in[4];   // [E,H]
    const float* W2 = (const float*)d_in[5];   // [E,H,D]
    const float* b2 = (const float*)d_in[6];   // [E,D]

    float* out   = (float*)d_out;                          // [B,T,D]
    float* probs = out + (size_t)BT_ * DD_;                // [B,T,E]
    int*   eidx  = (int*)(probs + (size_t)BT_ * EE_);      // [B,T]

    char*   ws     = (char*)d_ws;
    int*    counts = (int*)ws;
    int*    list   = (int*)(ws + LIST_OFF);
    __bf16* W1t    = (__bf16*)(ws + W1T_OFF);
    __bf16* W2t    = (__bf16*)(ws + W2T_OFF);
    __bf16* hbuf   = (__bf16*)(ws + HBUF_OFF);

    moe_zero_counts<<<1, 32, 0, stream>>>(counts);

    moe_cvt_transpose<<<dim3(HH_ / 32, DD_ / 32, EE_), 256, 0, stream>>>(W1, W1t, DD_, HH_);
    moe_cvt_transpose<<<dim3(DD_ / 32, HH_ / 32, EE_), 256, 0, stream>>>(W2, W2t, HH_, DD_);

    moe_router<<<BT_ / 8, 256, 0, stream>>>(x, Wr, br, probs, eidx, counts, list);

    moe_gemm<DD_, HH_, true ><<<dim3(HH_ / 128, BT_ / 128, EE_), 256, 0, stream>>>(
        x, W1t, b1, counts, list, hbuf);
    moe_gemm<HH_, DD_, false><<<dim3(DD_ / 128, BT_ / 128, EE_), 256, 0, stream>>>(
        hbuf, W2t, b2, counts, list, out);
}
